// GridV2_50165218017427
// MI455X (gfx1250) — matrix operand order repaired and checked by reading, not compiled, not run on hardware
//
#include <hip/hip_runtime.h>
#include <hip/hip_bf16.h>
#include <math.h>
#include <stdint.h>

typedef __attribute__((ext_vector_type(16))) _Float16 v16h;
typedef __attribute__((ext_vector_type(8)))  _Float16 v8h;
typedef __attribute__((ext_vector_type(8)))  float    v8f;

#define GSZ 32
#define L_  1024
#define B_  4
#define NN_ 20
#define C_  128
#define H_  8
#define W_  32
#define HW_ 256
#define K_  (NN_*HW_)   // 5120
#define KT_ (K_/32)     // 160 k-tiles of 32
#define NT_ (C_/16)     // 8 n-tiles of 16
#define MT_ (L_/16)     // 64 m-tiles of 16

// ---------------------------------------------------------------------------
// Kernel 1: pano[b,n,{max,mean},x,y] = max/mean over C of feats[b,n,:,x,y]
// ---------------------------------------------------------------------------
__global__ void prep_pano(const float* __restrict__ feats, float* __restrict__ pano) {
  int idx = blockIdx.x * 256 + threadIdx.x;          // [0, B*NN*HW)
  if (idx >= B_ * NN_ * HW_) return;
  int pix = idx & (HW_ - 1);
  int bn  = idx >> 8;                                // b*NN + n
  const float* p = feats + (size_t)bn * C_ * HW_ + pix;
  float mx = -INFINITY, sm = 0.f;
  #pragma unroll 4
  for (int c = 0; c < C_; ++c) { float v = p[c * HW_]; mx = fmaxf(mx, v); sm += v; }
  pano[(size_t)bn * 2 * HW_ + pix]        = mx;
  pano[(size_t)bn * 2 * HW_ + HW_ + pix]  = sm * (1.0f / C_);
}

// ---------------------------------------------------------------------------
// Kernel 2: ov{max,mean}[b,s] over channel dim of overhead[b,:,s]
// ---------------------------------------------------------------------------
__global__ void prep_ov(const float* __restrict__ over,
                        float* __restrict__ ovmax, float* __restrict__ ovmean) {
  int idx = blockIdx.x * 256 + threadIdx.x;          // [0, B*L)
  if (idx >= B_ * L_) return;
  int s = idx & (L_ - 1);
  int b = idx >> 10;
  const float* p = over + (size_t)b * C_ * L_ + s;
  float mx = -INFINITY, sm = 0.f;
  #pragma unroll 4
  for (int c = 0; c < C_; ++c) { float v = p[c * L_]; mx = fmaxf(mx, v); sm += v; }
  ovmax[idx]  = mx;
  ovmean[idx] = sm * (1.0f / C_);
}

// ---------------------------------------------------------------------------
// Kernel 3: build B matrix in f16, pre-swizzled to the WMMA B-fragment layout:
//   bsw[ ((b*NT+nt)*KT + kt)*32 + lane ][ j ]  (16 contiguous halfs per lane)
//   lane = h*16 + c holds K = kt*32 + 16*(j>>3) + h*8 + (j&7), column n = nt*16 + c
//   B[k, n] = feats[b, p, n, x, y],  k = p*256 + x*32 + y
// ---------------------------------------------------------------------------
__global__ void build_bsw(const float* __restrict__ feats, _Float16* __restrict__ bsw) {
  size_t idx = (size_t)blockIdx.x * 256 + threadIdx.x;
  const size_t total = (size_t)B_ * NT_ * KT_ * 32 * 16;
  if (idx >= total) return;
  int j    = (int)(idx & 15);
  int lane = (int)((idx >> 4) & 31);
  size_t t = idx >> 9;                 // (b*NT + nt)*KT + kt
  int kt = (int)(t % KT_); t /= KT_;
  int nt = (int)(t & (NT_ - 1));
  int b  = (int)(t >> 3);
  int k  = kt * 32 + (lane >> 4) * 8 + (j & 7) + (j >> 3) * 16;
  int n  = nt * 16 + (lane & 15);
  int p  = k >> 8;
  int rem = k & 255;
  float v = feats[(((size_t)(b * NN_ + p)) * C_ + n) * HW_ + rem];
  bsw[idx] = (_Float16)v;
}

// ---------------------------------------------------------------------------
// Kernel 4: one block per (b, grid cell s). Compute the 5120 logits, softmax
// in LDS, write attn weights (f32) to d_out. Ray/pano maps are staged into
// LDS with CDNA5 async global->LDS copies (ASYNCcnt / s_wait_asynccnt).
// ---------------------------------------------------------------------------
__global__ void __launch_bounds__(256)
logits_softmax(const float* __restrict__ bbox, const float* __restrict__ locs,
               const float* __restrict__ rays,
               const float* __restrict__ c1w, const float* __restrict__ c1b,
               const float* __restrict__ c2w, const float* __restrict__ c2b,
               const float* __restrict__ fw,  const float* __restrict__ fb,
               const float* __restrict__ pano,
               const float* __restrict__ ovmax, const float* __restrict__ ovmean,
               float* __restrict__ attn) {
  __shared__ float sM[5 * HW_];       // 3 ray maps + 2 pano maps for current n
  __shared__ float sL[NN_ * HW_];     // logits
  __shared__ float sD[NN_];
  __shared__ int   sI[NN_];
  __shared__ float sW1[72], sW2[200];
  __shared__ float sRed[256];

  const int tid = threadIdx.x;
  const int bs  = blockIdx.x;
  const int b   = bs >> 10;
  const int s   = bs & (L_ - 1);
  const int iy  = s >> 5, ix = s & 31;

  if (tid < 72)  sW1[tid] = c1w[tid];
  if (tid < 200) sW2[tid] = c2w[tid];
  if (tid < NN_) {
    float y0 = bbox[b * 4 + 0], x0 = bbox[b * 4 + 1];
    float y1 = bbox[b * 4 + 2], x1 = bbox[b * 4 + 3];
    float gy = y0 + (y1 - y0) * (float)iy * (1.0f / 31.0f);
    float gx = x0 + (x1 - x0) * (float)ix * (1.0f / 31.0f);
    float d0 = gy - locs[(b * NN_ + tid) * 2 + 0];
    float d1 = gx - locs[(b * NN_ + tid) * 2 + 1];
    sD[tid] = sqrtf(d0 * d0 + d1 * d1 + 1e-12f);
    float th = atan2f(d1, d0);
    if (th < 0.f) th += 6.283185307179586f;
    int deg = (int)roundf(th * 57.29577951308232f);
    deg %= 360; if (deg < 0) deg += 360;
    sI[tid] = deg;
  }
  __syncthreads();

  // kernel-weight sums for spatially-constant channels 0 (D), 6 (ovmax), 7 (ovmean)
  float s1_0 = 0, s1_6 = 0, s1_7 = 0, s2_0 = 0, s2_6 = 0, s2_7 = 0;
  #pragma unroll
  for (int i = 0; i < 9;  ++i) { s1_0 += sW1[i];      s1_6 += sW1[54 + i];  s1_7 += sW1[63 + i];  }
  #pragma unroll
  for (int i = 0; i < 25; ++i) { s2_0 += sW2[i];      s2_6 += sW2[150 + i]; s2_7 += sW2[175 + i]; }

  const float ox = ovmax[bs], om = ovmean[bs];
  const float b1 = c1b[0], b2 = c2b[0];
  const float f0 = fw[0], f1 = fw[1], fbb = fb[0];
  const int x = tid >> 5, y = tid & 31;

  for (int n = 0; n < NN_; ++n) {
    __syncthreads();                  // sM reads of previous iter complete
    {
      // Async global->LDS staging. The 24-bit inst offset is added to BOTH
      // the LDS and the global address; maps are 1KB apart on both sides.
      const float* r3 = rays + (size_t)sI[n] * 3 * HW_;
      const float* pp = pano + (size_t)(b * NN_ + n) * 2 * HW_;
      uint32_t lR = (uint32_t)(uintptr_t)(&sM[tid]);
      uint64_t gR = (uint64_t)(uintptr_t)(r3 + tid);
      asm volatile(
        "global_load_async_to_lds_b32 %0, %1, off\n\t"
        "global_load_async_to_lds_b32 %0, %1, off offset:1024\n\t"
        "global_load_async_to_lds_b32 %0, %1, off offset:2048"
        :: "v"(lR), "v"(gR) : "memory");
      uint32_t lP = (uint32_t)(uintptr_t)(&sM[3 * HW_ + tid]);
      uint64_t gP = (uint64_t)(uintptr_t)(pp + tid);
      asm volatile(
        "global_load_async_to_lds_b32 %0, %1, off\n\t"
        "global_load_async_to_lds_b32 %0, %1, off offset:1024"
        :: "v"(lP), "v"(gP) : "memory");
      asm volatile("s_wait_asynccnt 0x0" ::: "memory");
    }
    __syncthreads();
    float D  = sD[n];
    float c1 = b1 + D * s1_0 + ox * s1_6 + om * s1_7;
    float c2 = b2 + D * s2_0 + ox * s2_6 + om * s2_7;
    #pragma unroll
    for (int c = 1; c <= 5; ++c) {
      const float* m  = sM + (c - 1) * HW_;
      const float* w1 = sW1 + c * 9;
      #pragma unroll
      for (int ky = 0; ky < 3; ++ky) {
        int xx = (x + ky + 7) & 7;
        #pragma unroll
        for (int kx = 0; kx < 3; ++kx) {
          int yy = (y + kx + 31) & 31;
          c1 = fmaf(w1[ky * 3 + kx], m[xx * 32 + yy], c1);
        }
      }
      const float* w2 = sW2 + c * 25;
      #pragma unroll
      for (int ky = 0; ky < 5; ++ky) {
        int xx = (x + ky + 6) & 7;
        #pragma unroll
        for (int kx = 0; kx < 5; ++kx) {
          int yy = (y + kx + 30) & 31;
          c2 = fmaf(w2[ky * 5 + kx], m[xx * 32 + yy], c2);
        }
      }
    }
    sL[n * HW_ + tid] = fbb + f0 * c1 + f1 * c2;
  }
  __syncthreads();

  // block softmax over all 5120 logits
  float mx = -INFINITY;
  for (int n = 0; n < NN_; ++n) mx = fmaxf(mx, sL[n * HW_ + tid]);
  sRed[tid] = mx; __syncthreads();
  for (int off = 128; off > 0; off >>= 1) {
    if (tid < off) sRed[tid] = fmaxf(sRed[tid], sRed[tid + off]);
    __syncthreads();
  }
  mx = sRed[0]; __syncthreads();
  float sm = 0.f;
  for (int n = 0; n < NN_; ++n) sm += __expf(sL[n * HW_ + tid] - mx);
  sRed[tid] = sm; __syncthreads();
  for (int off = 128; off > 0; off >>= 1) {
    if (tid < off) sRed[tid] += sRed[tid + off];
    __syncthreads();
  }
  const float inv = 1.0f / sRed[0];

  float* ao = attn + (size_t)bs * K_;
  for (int n = 0; n < NN_; ++n)
    ao[n * HW_ + tid] = __expf(sL[n * HW_ + tid] - mx) * inv;
}

// ---------------------------------------------------------------------------
// Kernel 5: grid_feats[b,s,f] = sum_k attn[b,s,k] * B[k,f] via WMMA f16.
// One wave per 16x32 output strip (two adjacent 16x16 N-tiles): the A
// fragment (f32->f16 converted on load) is reused across both WMMAs.
// ---------------------------------------------------------------------------
__global__ void __launch_bounds__(128)
gemm_wmma(const float* __restrict__ attn, const _Float16* __restrict__ bsw,
          float* __restrict__ outg) {
  int wave = (blockIdx.x * 128 + threadIdx.x) >> 5;   // [0, B*MT*NT/2)
  int lane = threadIdx.x & 31;
  int np = wave & (NT_ / 2 - 1);                      // 0..3
  int t  = wave >> 2;
  int mt = t & (MT_ - 1);
  int b  = t >> 6;
  const int nt0 = np * 2;

  const int r = lane & 15;
  const int h = lane >> 4;

  const float*    Arow = attn + (size_t)(b * L_ + mt * 16 + r) * K_;
  const _Float16* B0   = bsw + (size_t)(b * NT_ + nt0)     * KT_ * 512;
  const _Float16* B1   = bsw + (size_t)(b * NT_ + nt0 + 1) * KT_ * 512;

  v8f acc0 = {}, acc1 = {};
  for (int kt = 0; kt < KT_; ++kt) {
    const float*    ap  = Arow + kt * 32 + h * 8;
    const _Float16* bp0 = B0 + kt * 512 + lane * 16;
    const _Float16* bp1 = B1 + kt * 512 + lane * 16;
    if (kt + 1 < KT_) {
      __builtin_prefetch(ap + 32, 0, 3);    // -> global_prefetch_b8
      __builtin_prefetch(bp0 + 512, 0, 3);
      __builtin_prefetch(bp1 + 512, 0, 3);
    }
    v8h b0lo = *(const v8h*)bp0;            // lane's K = h*8 + 0..7
    v8h b0hi = *(const v8h*)(bp0 + 8);      // lane's K = 16 + h*8 + 0..7
    v8h b1lo = *(const v8h*)bp1;
    v8h b1hi = *(const v8h*)(bp1 + 8);
    v16h a, bb0, bb1;
    #pragma unroll
    for (int i = 0; i < 8; ++i) {
      a[i]       = (_Float16)ap[i];         // K = h*8 + i       (VGPRs 0-3)
      a[i + 8]   = (_Float16)ap[16 + i];    // K = 16 + h*8 + i  (VGPRs 4-7)
      bb0[i]     = b0lo[i];
      bb0[i + 8] = b0hi[i];
      bb1[i]     = b1lo[i];
      bb1[i + 8] = b1hi[i];
    }
    acc0 = __builtin_amdgcn_wmma_f32_16x16x32_f16(
        false, a, false, bb0, (short)0, acc0, false, false);
    acc1 = __builtin_amdgcn_wmma_f32_16x16x32_f16(
        false, a, false, bb1, (short)0, acc1, false, false);
  }

  // D layout: VGPR v <-> row (v + 8*h), col = lane%16
  float* o0 = outg + (size_t)(b * L_ + mt * 16) * C_ + nt0 * 16 + (lane & 15);
  #pragma unroll
  for (int v = 0; v < 8; ++v) {
    o0[(size_t)(v + 8 * h) * C_]      = acc0[v];
    o0[(size_t)(v + 8 * h) * C_ + 16] = acc1[v];
  }
}

// ---------------------------------------------------------------------------
extern "C" void kernel_launch(void* const* d_in, const int* in_sizes, int n_in,
                              void* d_out, int out_size, void* d_ws, size_t ws_size,
                              hipStream_t stream) {
  (void)in_sizes; (void)n_in; (void)out_size; (void)ws_size;
  const float* bbox  = (const float*)d_in[0];
  const float* locs  = (const float*)d_in[1];
  const float* feats = (const float*)d_in[2];
  const float* over  = (const float*)d_in[3];
  const float* rays  = (const float*)d_in[4];
  const float* c1w   = (const float*)d_in[5];
  const float* c1b   = (const float*)d_in[6];
  const float* c2w   = (const float*)d_in[7];
  const float* c2b   = (const float*)d_in[8];
  const float* fw    = (const float*)d_in[9];
  const float* fb    = (const float*)d_in[10];

  float* outg = (float*)d_out;                       // grid_feats [B, L, C]
  float* attn = outg + (size_t)B_ * L_ * C_;         // attn       [B, L, K]

  // workspace layout (bytes):
  //   pano  f32 : B*NN*2*HW        = 163840
  //   ovmax f32 : B*L              =  16384
  //   ovmean f32: B*L              =  16384
  //   bsw   f16 : B*K*C            = 5242880   (total ~5.4 MB)
  char* ws = (char*)d_ws;
  float*    pano   = (float*)ws;
  float*    ovmax  = (float*)(ws + 163840);
  float*    ovmean = (float*)(ws + 180224);
  _Float16* bsw    = (_Float16*)(ws + 196608);

  prep_pano<<<(B_ * NN_ * HW_ + 255) / 256, 256, 0, stream>>>(feats, pano);
  prep_ov<<<(B_ * L_ + 255) / 256, 256, 0, stream>>>(over, ovmax, ovmean);
  build_bsw<<<(int)(((size_t)B_ * NT_ * KT_ * 512 + 255) / 256), 256, 0, stream>>>(feats, bsw);
  logits_softmax<<<B_ * L_, 256, 0, stream>>>(bbox, locs, rays, c1w, c1b, c2w, c2b,
                                              fw, fb, pano, ovmax, ovmean, attn);
  gemm_wmma<<<(B_ * MT_ * (NT_ / 2)) / 4, 128, 0, stream>>>(attn, bsw, outg);
}
